// DistiledMultiheadAttention_76476187673064
// MI455X (gfx1250) — matrix operand order repaired and checked by reading, not compiled, not run on hardware
//
#include <hip/hip_runtime.h>

typedef _Float16 half_t;
typedef __attribute__((ext_vector_type(2)))  _Float16 h2;
typedef __attribute__((ext_vector_type(16))) _Float16 v16h;
typedef __attribute__((ext_vector_type(8)))  float    v8f;

#define MT 128
#define NT 64
#define KT 32
#define LDSK 40   // padded LDS row stride (halfs); 80B rows keep 16B-aligned b128 accesses

// ---------------- CDNA5 async-copy helpers (ASYNCcnt path) ----------------
// global_load_async_to_lds_b128: per-lane 16B copy global -> LDS, tracked by ASYNCcnt.
__device__ __forceinline__ void async_copy16(const half_t* gptr, half_t* lptr) {
  unsigned lds = (unsigned)(uintptr_t)lptr;                 // low 32 bits = LDS byte address
  unsigned long long ga = (unsigned long long)(uintptr_t)gptr;
  asm volatile("global_load_async_to_lds_b128 %0, %1, off"
               :: "v"(lds), "v"(ga) : "memory");
}
#define WAIT_ASYNC(n) asm volatile("s_wait_asynccnt " #n ::: "memory")

// ---------------- prep kernels ----------------

__global__ void zero_f32_v4(float* __restrict__ p, int n4) {
  int i = blockIdx.x * blockDim.x + threadIdx.x;
  if (i < n4) ((float4*)p)[i] = make_float4(0.f, 0.f, 0.f, 0.f);
}

__global__ void f32_to_f16_v4(const float* __restrict__ src, half_t* __restrict__ dst, int n4) {
  int i = blockIdx.x * blockDim.x + threadIdx.x;
  if (i >= n4) return;
  float4 v = ((const float4*)src)[i];
  h2 a = { (half_t)v.x, (half_t)v.y };
  h2 b = { (half_t)v.z, (half_t)v.w };
  *(h2*)(dst + 4 * (size_t)i)     = a;
  *(h2*)(dst + 4 * (size_t)i + 2) = b;
}

// per-token segment-local index j and padded output row seg*MAXL + j
__global__ void build_maps(const int* __restrict__ lengths, int* __restrict__ row_map,
                           int* __restrict__ jmap, int T, int Bn, int maxl) {
  int t = blockIdx.x * blockDim.x + threadIdx.x;
  if (t >= T) return;
  int start = 0, seg = 0;
  for (int s = 0; s < Bn; ++s) {
    int len = lengths[s];
    if (t < start + len) { seg = s; break; }
    start += len;
  }
  int j = t - start;
  row_map[t] = seg * maxl + j;
  jmap[t] = j;
}

// ---------------- WMMA GEMM: C[M,N] = A[M,Kd] (f16) x W[N,Kd]^T (f16) + bias ----------------
// Both operands are K-contiguous row-major, so A tiles AND W tiles stage into LDS with
// straight async b128 copies (no transpose anywhere). Double-buffered LDS; each thread
// issues exactly 3 async copies per K-step (2 for the 128x32 A tile, 1 for the 64x32 W tile).
// block = 256 threads (8 wave32), block tile 128x64, wave tile 32x32 (2x2 of 16x16 WMMA).

template <bool HAS_MAP>
__global__ __launch_bounds__(256)
void gemm_xwt_wmma(const half_t* __restrict__ A, const half_t* __restrict__ Wm,
                   const float* __restrict__ bias, float* __restrict__ C,
                   int M, int N, int Kd, const int* __restrict__ row_map) {
  __shared__ half_t As[2][MT * LDSK];   // [buf][128][40]
  __shared__ half_t Ws[2][NT * LDSK];   // [buf][64][40]   (row n of W, K-contiguous)

  const int tid  = threadIdx.x;
  const int lane = tid & 31;
  const int wv   = tid >> 5;        // 0..7
  const int wm   = wv >> 1;         // 0..3 (M dir)
  const int wn   = wv & 1;          // 0..1 (N dir)
  const int m0   = blockIdx.y * MT;
  const int n0   = blockIdx.x * NT;
  const int lane15   = lane & 15;
  const int laneHalf = lane >> 4;   // 0 or 1

  // per-thread staging coordinates: 64 rows x 4 chunks of 8 halfs
  const int sr = tid >> 2;          // 0..63
  const int sc = (tid & 3) * 8;     // 0,8,16,24

  v8f acc[2][2] = {};

  const int NK = Kd / KT;

  // prologue prefetch of tile 0 into buffer 0
  {
    const half_t* ga = A  + (size_t)(m0 + sr) * Kd + sc;
    const half_t* gb = Wm + (size_t)(n0 + sr) * Kd + sc;
    async_copy16(ga,                    &As[0][sr * LDSK + sc]);
    async_copy16(ga + (size_t)64 * Kd,  &As[0][(sr + 64) * LDSK + sc]);
    async_copy16(gb,                    &Ws[0][sr * LDSK + sc]);
  }

  for (int kt = 0; kt < NK; ++kt) {
    const int cur = kt & 1;
    if (kt + 1 < NK) {
      const int k0 = (kt + 1) * KT;
      const half_t* ga = A  + (size_t)(m0 + sr) * Kd + k0 + sc;
      const half_t* gb = Wm + (size_t)(n0 + sr) * Kd + k0 + sc;
      async_copy16(ga,                   &As[1 - cur][sr * LDSK + sc]);
      async_copy16(ga + (size_t)64 * Kd, &As[1 - cur][(sr + 64) * LDSK + sc]);
      async_copy16(gb,                   &Ws[1 - cur][sr * LDSK + sc]);
      WAIT_ASYNC(3);     // tile kt's 3 copies done; kt+1's 3 still in flight
    } else {
      WAIT_ASYNC(0);
    }
    __syncthreads();

    // A fragments (16x32): lane<16 -> K {0..7,16..23}, lane>=16 -> K {8..15,24..31}
    v16h afrag[2], bfrag[2];
#pragma unroll
    for (int i = 0; i < 2; ++i) {
      const half_t* ap = &As[cur][(wm * 32 + i * 16 + lane15) * LDSK + laneHalf * 8];
#pragma unroll
      for (int p = 0; p < 4; ++p) {
        h2 lo = *(const h2*)(ap + 2 * p);
        h2 hi = *(const h2*)(ap + 16 + 2 * p);
        afrag[i][2 * p]     = lo.x; afrag[i][2 * p + 1] = lo.y;
        afrag[i][8 + 2 * p] = hi.x; afrag[i][9 + 2 * p] = hi.y;
      }
    }
    // B fragments (32x16): lane holds column n = row n of W, K {0..7,16..23} / {8..15,24..31}
#pragma unroll
    for (int jb = 0; jb < 2; ++jb) {
      const half_t* bp = &Ws[cur][(wn * 32 + jb * 16 + lane15) * LDSK + laneHalf * 8];
#pragma unroll
      for (int p = 0; p < 4; ++p) {
        h2 lo = *(const h2*)(bp + 2 * p);
        h2 hi = *(const h2*)(bp + 16 + 2 * p);
        bfrag[jb][2 * p] = lo.x; bfrag[jb][2 * p + 1] = lo.y;
        bfrag[jb][8 + 2 * p] = hi.x; bfrag[jb][9 + 2 * p] = hi.y;
      }
    }

#pragma unroll
    for (int i = 0; i < 2; ++i)
#pragma unroll
      for (int jb = 0; jb < 2; ++jb)
        acc[i][jb] = __builtin_amdgcn_wmma_f32_16x16x32_f16(
            false, afrag[i], false, bfrag[jb], (short)0, acc[i][jb], false, false);
    __syncthreads();
  }

  // epilogue: bias add + optional compile-time row scatter (pad_sequence)
  const int nc0 = n0 + wn * 32 + lane15;
  const float bv0 = bias[nc0];
  const float bv1 = bias[nc0 + 16];
#pragma unroll
  for (int i = 0; i < 2; ++i) {
#pragma unroll
    for (int vg = 0; vg < 8; ++vg) {
      const int mrow = m0 + wm * 32 + i * 16 + vg + laneHalf * 8;
      const int orow = HAS_MAP ? row_map[mrow] : mrow;
      float* crow = C + (size_t)orow * N;
      crow[nc0]      = acc[i][0][vg] + bv0;
      crow[nc0 + 16] = acc[i][1][vg] + bv1;
    }
  }
}

// ---------------- windowed attention core ----------------
// one wave32 per (token t, head h); lane = window slot w (0..31).
// kv[T,1536]: cols [0,512) keys (h*32+d), cols [512,1536) values (512+h*64+d).
// window source row: kv[t-w] if j>=w else buffer[j-w+31]. kv is L2-resident (24 MB).

__global__ __launch_bounds__(256)
void attn_window(const float* __restrict__ q, const float* __restrict__ kv,
                 const float* __restrict__ buffer, const int* __restrict__ jmap,
                 half_t* __restrict__ oh) {
  const int lane = threadIdx.x & 31;
  const int wv   = threadIdx.x >> 5;
  const int pair = blockIdx.x * 8 + wv;   // t*H + h
  const int t = pair >> 4;
  const int h = pair & 15;
  const int j = jmap[t];

  const float* srcw = (j >= lane) ? (kv + (size_t)(t - lane) * 1536)
                                  : (buffer + (size_t)(j - lane + 31) * 1536);
  const float* qp = q + (size_t)t * 512 + h * 32;
  const float* kp = srcw + h * 32;
  float logit = 0.f;
#pragma unroll
  for (int d = 0; d < 32; ++d) logit += qp[d] * kp[d];
  logit *= 0.1767766953f;   // 1/sqrt(32)

  // softmax across the 32 lanes
  float mx = logit;
#pragma unroll
  for (int s = 16; s > 0; s >>= 1) mx = fmaxf(mx, __shfl_xor(mx, s, 32));
  float e = __expf(logit - mx);
  float sum = e;
#pragma unroll
  for (int s = 16; s > 0; s >>= 1) sum += __shfl_xor(sum, s, 32);
  float p = e / sum;

  // o[h*64 + d] = sum_w p[w] * v_w[d]; lane covers d and d+32
  float o0 = 0.f, o1 = 0.f;
  for (int w = 0; w < 32; ++w) {
    float pw = __shfl(p, w, 32);
    const float* sp = (j >= w) ? (kv + (size_t)(t - w) * 1536)
                               : (buffer + (size_t)(j - w + 31) * 1536);
    const float* vp = sp + 512 + h * 64;
    o0 += pw * vp[lane];
    o1 += pw * vp[lane + 32];
  }
  half_t* op = oh + (size_t)t * 1024 + h * 64;
  op[lane]      = (half_t)o0;
  op[lane + 32] = (half_t)o1;
}

// ---------------- launcher ----------------

extern "C" void kernel_launch(void* const* d_in, const int* in_sizes, int n_in,
                              void* d_out, int out_size, void* d_ws, size_t ws_size,
                              hipStream_t stream) {
  (void)in_sizes; (void)n_in; (void)ws_size;
  const float* x      = (const float*)d_in[0];
  const float* Wkv    = (const float*)d_in[1];
  const float* bkv    = (const float*)d_in[2];
  const float* Wq     = (const float*)d_in[3];
  const float* bq     = (const float*)d_in[4];
  const float* Wp     = (const float*)d_in[5];
  const float* bp     = (const float*)d_in[6];
  const float* buffer = (const float*)d_in[7];
  const int*   lens   = (const int*)d_in[8];
  float* out = (float*)d_out;

  const int T = 4096, E = 1024, K = 512, H = 16, Bn = 8, MAXL = 768;
  const int EK = E + K;  // 1536

  char* ws = (char*)d_ws;
  size_t off = 0;
  auto alloc = [&](size_t bytes) -> void* {
    void* p = ws + off;
    off += (bytes + 255) & ~(size_t)255;
    return p;
  };
  half_t* xh   = (half_t*)alloc((size_t)T * E * 2);     // x in f16        [T,E]
  half_t* Wkvh = (half_t*)alloc((size_t)EK * E * 2);    // Wkv f16 (as-is) [EK,E]
  half_t* Wqh  = (half_t*)alloc((size_t)K * E * 2);     // Wq  f16 (as-is) [K,E]
  half_t* Wph  = (half_t*)alloc((size_t)E * E * 2);     // Wp  f16 (as-is) [E,E]
  float*  kv   = (float*)alloc((size_t)T * EK * 4);     // kv projection f32
  float*  qf   = (float*)alloc((size_t)T * K * 4);      // q projection f32
  half_t* oh   = (half_t*)alloc((size_t)T * E * 2);     // attention out f16
  int* row_map = (int*)alloc((size_t)T * 4);
  int* jmap    = (int*)alloc((size_t)T * 4);

  // prep (all element counts divisible by 4)
  zero_f32_v4<<<(out_size / 4 + 255) / 256, 256, 0, stream>>>(out, out_size / 4);
  f32_to_f16_v4<<<(T * E / 4 + 255) / 256, 256, 0, stream>>>(x, xh, T * E / 4);
  f32_to_f16_v4<<<(EK * E / 4 + 255) / 256, 256, 0, stream>>>(Wkv, Wkvh, EK * E / 4);
  f32_to_f16_v4<<<(K * E / 4 + 255) / 256, 256, 0, stream>>>(Wq, Wqh, K * E / 4);
  f32_to_f16_v4<<<(E * E / 4 + 255) / 256, 256, 0, stream>>>(Wp, Wph, E * E / 4);
  build_maps<<<(T + 255) / 256, 256, 0, stream>>>(lens, row_map, jmap, T, Bn, MAXL);

  // kv = x @ Wkv^T + bkv   [4096,1536]
  dim3 g1(EK / NT, T / MT);
  gemm_xwt_wmma<false><<<g1, 256, 0, stream>>>(xh, Wkvh, bkv, kv, T, EK, E, nullptr);
  // q = x @ Wq^T + bq      [4096,512]
  dim3 g2(K / NT, T / MT);
  gemm_xwt_wmma<false><<<g2, 256, 0, stream>>>(xh, Wqh, bq, qf, T, K, E, nullptr);
  // windowed attention -> oh (f16)
  attn_window<<<T * H / 8, 256, 0, stream>>>(qf, kv, buffer, jmap, oh);
  // out[row_map[t]] = oh @ Wp^T + bp, scattered into zero-padded [B*MAXL, E]
  dim3 g3(E / NT, T / MT);
  gemm_xwt_wmma<true><<<g3, 256, 0, stream>>>(oh, Wph, bp, out, T, E, E, row_map);
}
